// QLSTMHybrid_65481071404485
// MI455X (gfx1250) — matrix operand order repaired and checked by reading, not compiled, not run on hardware
//
#include <hip/hip_runtime.h>
#include <hip/hip_bf16.h>

typedef __attribute__((ext_vector_type(16))) _Float16 v16h;
typedef __attribute__((ext_vector_type(8)))  float    v8f;

constexpr int S  = 512;
constexpr int B  = 2048;
constexpr int D  = 128;
constexpr int H  = 8;
constexpr int K3 = 3 * D;        // GEMM K = 384
constexpr int KCH = K3 / 32;     // 12 WMMA chunks of K=32
constexpr int WPACK_ELEMS = KCH * 512;   // 6144 halfs = 12 KB
constexpr float EPS = 1e-5f;

// ---------------------------------------------------------------------------
// Kernel 0 (one-shot): pack conv weights into WMMA B-fragment order, f16.
//   B (32x16, 16-bit): lanes 0-15 hold N=lane, K=0..7,16..23 (halfs 0..15);
//                      lanes 16-31 hold N=lane-16, K=8..15,24..31.
// wpack[c*512 + lane*16 + j] so each wave per chunk does 2x global_load_b128.
// ---------------------------------------------------------------------------
__global__ __launch_bounds__(256) void qlstm_pack_w(
    const float* __restrict__ w,        // [H, D, 3]
    _Float16* __restrict__ wpack)       // [KCH * 512]
{
  int idx = blockIdx.x * blockDim.x + threadIdx.x;
  if (idx >= WPACK_ELEMS) return;
  int c   = idx >> 9;                   // chunk
  int rem = idx & 511;
  int l   = rem >> 4;                   // lane
  int j   = rem & 15;                   // half index within lane
  int n   = l & 15;                     // output column
  int kl  = ((l >> 4) << 3) + ((j >> 3) << 4) + (j & 7);  // lanes>=16 hold K 8..15, 24..31 of the chunk
  int kg  = c * 32 + kl;                // global GEMM K in [0, 384)
  int kk  = kg >> 7;                    // conv tap 0..2
  int d   = kg & 127;                   // input channel
  wpack[idx] = (_Float16)((n < H) ? w[n * K3 + d * 3 + kk] : 0.0f);
}

// ---------------------------------------------------------------------------
// Kernel 1: Conv1d(k=3, pad=SAME, no bias) as im2col GEMM via WMMA f16->f32.
//   enc[s,b,h] = sum_{kk,d} in[s+kk-1, b, d] * w[h, d, kk]
// One wave per (s, 16-batch-row) tile: A = X rows (16 x 384), B = packed
// weights (384 x 16, cols 8..15 zero), 12 x v_wmma_f32_16x16x32_f16.
// A fragment (16-bit A 16x32): lane<16: M=lane, halfs 0..7 = K 0..7,
//   halfs 8..15 = K 16..23; lane>=16: M=lane-16, K 8..15 and 24..31.
// No LDS, no barrier -> max resident waves; B fragments hit WGP$/L2.
// ---------------------------------------------------------------------------
__global__ __launch_bounds__(256) void qlstm_conv_wmma(
    const float* __restrict__ in,       // [S, B, D]
    const _Float16* __restrict__ wpack, // [KCH * 512] fragment-ordered
    float* __restrict__ enc)            // [S, B, H]
{
  const int wave = threadIdx.x >> 5;
  const int lane = threadIdx.x & 31;
  const int tile = blockIdx.x * 8 + wave;   // tile = bt*S + s (s inner -> L2 reuse)
  const int s    = tile & (S - 1);
  const int bt   = tile >> 9;               // tile / 512
  const int b0   = bt * 16;
  const int row  = lane & 15;
  const int hi   = lane >> 4;

  v8f acc = {0.f, 0.f, 0.f, 0.f, 0.f, 0.f, 0.f, 0.f};

#pragma unroll
  for (int c = 0; c < KCH; ++c) {
    v16h a;
    const int base0 = c * 32 + hi * 8;
#pragma unroll
    for (int seg = 0; seg < 2; ++seg) {
      const int kg0 = base0 + seg * 16;     // first K of this 8-wide segment
      const int kk  = kg0 >> 7;             // conv tap (segment never straddles)
      const int d0  = kg0 & 127;
      const int ss  = s + kk - 1;
      float4 f0, f1;
      if (ss >= 0 && ss < S) {
        const float* p = in + ((size_t)ss * B + b0 + row) * D + d0;
        f0 = *(const float4*)p;
        f1 = *(const float4*)(p + 4);
      } else {
        f0 = make_float4(0.f, 0.f, 0.f, 0.f);
        f1 = f0;
      }
      const int o = seg * 8;
      a[o + 0] = (_Float16)f0.x;  a[o + 1] = (_Float16)f0.y;
      a[o + 2] = (_Float16)f0.z;  a[o + 3] = (_Float16)f0.w;
      a[o + 4] = (_Float16)f1.x;  a[o + 5] = (_Float16)f1.y;
      a[o + 6] = (_Float16)f1.z;  a[o + 7] = (_Float16)f1.w;
    }
    v16h bfr = *(const v16h*)(wpack + c * 512 + lane * 16);
    acc = __builtin_amdgcn_wmma_f32_16x16x32_f16(
        /*neg_a=*/false, a, /*neg_b=*/false, bfr,
        /*c_mod=*/(short)0, acc, /*reuse_a=*/false, /*reuse_b=*/false);
  }

  // C/D layout: lane l: N = l&15, M = r + 8*(l>>4) for VGPR r.
  const int n = lane & 15;
  if (n < H) {
#pragma unroll
    for (int r = 0; r < 8; ++r) {
      const int m = r + hi * 8;
      enc[((size_t)s * B + b0 + m) * H + n] = acc[r];
    }
  }
}

// ---------------------------------------------------------------------------
// Kernel 2: the 512-step QLSTM recurrence. Each batch row is independent;
// 8 lanes per row (one per hidden unit), 4 rows per wave32.
// ---------------------------------------------------------------------------
__device__ __forceinline__ float qexp8(float theta, int q) {
  // <Z_q> of RY-ladder + CNOT ring: out0 = prod_{1..7} cos, out_q = prod_{0..q} cos.
  float c = __cosf(theta);
  float pp = (q == 0) ? 1.0f : c;                 // exclude c0 from the scan
  float u;
  u = __shfl_up(pp, 1, 8); if (q >= 1) pp *= u;
  u = __shfl_up(pp, 2, 8); if (q >= 2) pp *= u;
  u = __shfl_up(pp, 4, 8); if (q >= 4) pp *= u;   // pp[q] = prod_{1..q} c
  float all = __shfl(pp, 7, 8);                   // prod_{1..7} c
  float c0  = __shfl(c, 0, 8);
  return (q == 0) ? all : c0 * pp;
}

__device__ __forceinline__ float fast_sigmoid(float x) {
  return 1.0f / (1.0f + __expf(-x));
}
__device__ __forceinline__ float fast_tanh(float x) {
  return 1.0f - 2.0f / (__expf(2.0f * x) + 1.0f);
}

__global__ __launch_bounds__(256) void qlstm_scan(
    const float* __restrict__ enc,    // [S, B, H]
    const float* __restrict__ ln_g, const float* __restrict__ ln_b,
    const float* __restrict__ Wf, const float* __restrict__ bf,
    const float* __restrict__ Wi, const float* __restrict__ bi,
    const float* __restrict__ Wg, const float* __restrict__ bg,
    const float* __restrict__ Wo, const float* __restrict__ bo,
    float* __restrict__ outp,         // [S, B, H]
    float* __restrict__ hx_out,       // [B, H]
    float* __restrict__ cx_out)       // [B, H]
{
  const int tid = blockIdx.x * blockDim.x + threadIdx.x;
  const int q = tid & 7;              // hidden unit
  const int b = tid >> 3;             // batch row
  if (b >= B) return;

  // Per-lane gate weight rows (uniform across rows -> L2/WGP$ broadcast).
  float wf[16], wii[16], wg[16], wo[16];
#pragma unroll
  for (int j = 0; j < 16; ++j) {
    wf[j]  = Wf[q * 16 + j];
    wii[j] = Wi[q * 16 + j];
    wg[j]  = Wg[q * 16 + j];
    wo[j]  = Wo[q * 16 + j];
  }
  const float bfq = bf[q], biq = bi[q], bgq = bg[q], boq = bo[q];
  const float gx = ln_g[q],     bxx = ln_b[q];
  const float gh = ln_g[8 + q], bhh = ln_b[8 + q];

  float hx = 0.0f, cx = 0.0f;

  for (int s = 0; s < S; ++s) {
    const size_t off = ((size_t)s * B + b) * H + q;
    const float x = enc[off];
    if (s + 1 < S) __builtin_prefetch(enc + off + (size_t)B * H, 0, 3);

    // LayerNorm over the 16 features of this row.
    float sum = x + hx;
    sum += __shfl_xor(sum, 1, 8);
    sum += __shfl_xor(sum, 2, 8);
    sum += __shfl_xor(sum, 4, 8);
    const float mu = sum * (1.0f / 16.0f);
    const float dx = x - mu, dh = hx - mu;
    float sq = dx * dx + dh * dh;
    sq += __shfl_xor(sq, 1, 8);
    sq += __shfl_xor(sq, 2, 8);
    sq += __shfl_xor(sq, 4, 8);
    const float rstd = rsqrtf(sq * (1.0f / 16.0f) + EPS);
    const float nx = dx * rstd * gx + bxx;   // comb[q]
    const float nh = dh * rstd * gh + bhh;   // comb[8+q]

    // Gate pre-activations: theta_g[q] = b_g[q] + sum_j comb[j] * Wg[q,j]
    float tf = bfq, ti = biq, tg = bgq, to = boq;
#pragma unroll
    for (int t = 0; t < 8; ++t) {
      const float vx = __shfl(nx, t, 8);
      const float vh = __shfl(nh, t, 8);
      tf += vx * wf[t]  + vh * wf[8 + t];
      ti += vx * wii[t] + vh * wii[8 + t];
      tg += vx * wg[t]  + vh * wg[8 + t];
      to += vx * wo[t]  + vh * wo[8 + t];
    }

    const float f = fast_sigmoid(qexp8(tf, q));
    const float i = fast_sigmoid(qexp8(ti, q));
    const float g = fast_tanh  (qexp8(tg, q));
    const float o = fast_sigmoid(qexp8(to, q));

    cx = f * cx + i * g;
    hx = o * fast_tanh(cx);

    outp[off] = hx;
  }

  hx_out[b * H + q] = hx;
  cx_out[b * H + q] = cx;
}

// ---------------------------------------------------------------------------
extern "C" void kernel_launch(void* const* d_in, const int* in_sizes, int n_in,
                              void* d_out, int out_size, void* d_ws, size_t ws_size,
                              hipStream_t stream) {
  const float* inputs = (const float*)d_in[0];
  const float* conv_w = (const float*)d_in[1];
  const float* ln_g   = (const float*)d_in[2];
  const float* ln_b   = (const float*)d_in[3];
  const float* Wf = (const float*)d_in[4];  const float* bf = (const float*)d_in[5];
  const float* Wi = (const float*)d_in[6];  const float* bi = (const float*)d_in[7];
  const float* Wg = (const float*)d_in[8];  const float* bg = (const float*)d_in[9];
  const float* Wo = (const float*)d_in[10]; const float* bo = (const float*)d_in[11];

  float* outp   = (float*)d_out;                 // [S, B, H]
  float* hx_out = outp + (size_t)S * B * H;      // [B, H]
  float* cx_out = hx_out + (size_t)B * H;        // [B, H]

  // Workspace: [0, 16KB) packed f16 weights; [16KB, 16KB+32MB) enc.
  _Float16* wpack = (_Float16*)d_ws;
  float*    enc   = (float*)((char*)d_ws + 16384);

  // One-shot weight repack into WMMA B-fragment order.
  qlstm_pack_w<<<(WPACK_ELEMS + 255) / 256, 256, 0, stream>>>(conv_w, wpack);

  // Conv encoder: S*(B/16) tiles, 8 waves (tiles) per 256-thread block.
  const int conv_tiles  = S * (B / 16);          // 65536
  qlstm_conv_wmma<<<conv_tiles / 8, 256, 0, stream>>>(inputs, wpack, enc);

  // Recurrent scan: one lane per (batch row, hidden unit).
  const int scan_threads = B * H;                // 16384
  qlstm_scan<<<scan_threads / 256, 256, 0, stream>>>(
      enc, ln_g, ln_b, Wf, bf, Wi, bi, Wg, bg, Wo, bo, outp, hx_out, cx_out);
}
